// SelfAttention_38938173506195
// MI455X (gfx1250) — matrix operand order repaired and checked by reading, not compiled, not run on hardware
//
#include <hip/hip_runtime.h>

// ---------------------------------------------------------------------------
// Self-attention forward for MI455X (gfx1250, wave32, WMMA bf16 16x16x32).
// Pipeline: f32->bf16 convert, QKV GEMM (WMMA), flash attention (WMMA +
// online softmax + async V staging into LDS), output GEMM (WMMA).
// ---------------------------------------------------------------------------

typedef unsigned short u16;
typedef __attribute__((ext_vector_type(16))) __bf16 v16bf;
typedef __attribute__((ext_vector_type(8)))  float  v8f;

#define Bb 2
#define Ss 2048
#define Dd 1024
#define Hh 16
#define DH 64

union BFrag {
  v16bf v;
  uint4 q[2];   // 32 bytes = 16 bf16 elements
};

__device__ __forceinline__ u16 f32_to_bf16(float f) {
  unsigned int u = __float_as_uint(f);
  u += 0x7FFFu + ((u >> 16) & 1u);   // round-to-nearest-even
  return (u16)(u >> 16);
}

__device__ __forceinline__ v8f wmma_bf16(v16bf a, v16bf b, v8f c) {
  // (neg_a, A, neg_b, B, c_mod, C, reuse_a, reuse_b)
  return __builtin_amdgcn_wmma_f32_16x16x32_bf16(false, a, false, b,
                                                 (short)0, c, false, false);
}

// reductions across the 16 lanes of each half-wave (C/D row lives there)
__device__ __forceinline__ float hmax16(float v) {
  v = fmaxf(v, __shfl_xor(v, 1, 32));
  v = fmaxf(v, __shfl_xor(v, 2, 32));
  v = fmaxf(v, __shfl_xor(v, 4, 32));
  v = fmaxf(v, __shfl_xor(v, 8, 32));
  return v;
}
__device__ __forceinline__ float hsum16(float v) {
  v += __shfl_xor(v, 1, 32);
  v += __shfl_xor(v, 2, 32);
  v += __shfl_xor(v, 4, 32);
  v += __shfl_xor(v, 8, 32);
  return v;
}

// async copy of 16 bytes: global -> LDS (gfx1250, tracked by ASYNCcnt)
__device__ __forceinline__ void async_g2l_b128(unsigned lds_addr,
                                               const void* gaddr) {
  asm volatile("global_load_async_to_lds_b128 %0, %1, off"
               :: "v"(lds_addr), "v"(gaddr) : "memory");
}
__device__ __forceinline__ void wait_async0() {
  asm volatile("s_wait_asynccnt 0" ::: "memory");
}

// ---------------------------------------------------------------------------
// f32 -> bf16 bulk convert (4-wide)
// ---------------------------------------------------------------------------
__global__ void cvt_bf16_kernel(const float4* __restrict__ s,
                                ushort4* __restrict__ d, int n4) {
  int i = blockIdx.x * blockDim.x + threadIdx.x;
  int stride = blockDim.x * gridDim.x;
  for (; i < n4; i += stride) {
    float4 f = s[i];
    ushort4 o;
    o.x = f32_to_bf16(f.x); o.y = f32_to_bf16(f.y);
    o.z = f32_to_bf16(f.z); o.w = f32_to_bf16(f.w);
    d[i] = o;
  }
}

// ---------------------------------------------------------------------------
// QKV projection: qkv = x @ w_in^T + b_in.
// One wave computes a 32(M) x 64(N) strip. Stores route q,k into head-major
// [b,h,s,dh] and v transposed into [b,h,dh,s] (for contiguous V B-fragments).
// ---------------------------------------------------------------------------
__global__ void gemm_qkv_kernel(const u16* __restrict__ xb,
                                const u16* __restrict__ wib,
                                const float* __restrict__ b_in,
                                u16* __restrict__ qh, u16* __restrict__ kh,
                                u16* __restrict__ vt) {
  const int lane = threadIdx.x & 31;
  const int n16  = lane & 15;
  const int hf   = lane >> 4;
  const int gw   = (blockIdx.x * blockDim.x + threadIdx.x) >> 5;
  const int mstrip = gw & 127;        // 128 strips of 32 rows (M=4096)
  const int nstrip = gw >> 7;         // 48 strips of 64 cols (N=3072)
  const int m0 = mstrip * 32, n0 = nstrip * 64;

  v8f acc[2][4] = {};
#pragma unroll 4
  for (int k0 = 0; k0 < Dd; k0 += 32) {
    BFrag a0, a1;
    const u16* ap = xb + (size_t)(m0 + n16) * Dd + k0;
    a0.q[0] = *(const uint4*)(ap + hf * 8);
    a0.q[1] = *(const uint4*)(ap + 16 + hf * 8);
    const u16* ap1 = ap + (size_t)16 * Dd;
    a1.q[0] = *(const uint4*)(ap1 + hf * 8);
    a1.q[1] = *(const uint4*)(ap1 + 16 + hf * 8);
#pragma unroll
    for (int nt = 0; nt < 4; ++nt) {
      BFrag bf;
      const u16* bp = wib + (size_t)(n0 + nt * 16 + n16) * Dd + k0;
      bf.q[0] = *(const uint4*)(bp + hf * 8);
      bf.q[1] = *(const uint4*)(bp + 16 + hf * 8);
      acc[0][nt] = wmma_bf16(a0.v, bf.v, acc[0][nt]);
      acc[1][nt] = wmma_bf16(a1.v, bf.v, acc[1][nt]);
    }
  }

#pragma unroll
  for (int mi = 0; mi < 2; ++mi)
#pragma unroll
    for (int nt = 0; nt < 4; ++nt) {
      const int col = n0 + nt * 16 + n16;
      const float bias = b_in[col];
#pragma unroll
      for (int r = 0; r < 8; ++r) {
        const int row = m0 + mi * 16 + r + 8 * hf;
        const u16 bv = f32_to_bf16(acc[mi][nt][r] + bias);
        const int bb = row >> 11;      // / S
        const int ss = row & (Ss - 1);
        if (col < Dd) {
          const int hh = col >> 6, dd = col & 63;
          qh[((((size_t)bb * Hh + hh) * Ss + ss) << 6) + dd] = bv;
        } else if (col < 2 * Dd) {
          const int c = col - Dd, hh = c >> 6, dd = c & 63;
          kh[((((size_t)bb * Hh + hh) * Ss + ss) << 6) + dd] = bv;
        } else {
          const int c = col - 2 * Dd, hh = c >> 6, dd = c & 63;
          vt[(((size_t)bb * Hh + hh) * DH + dd) * Ss + ss] = bv;
        }
      }
    }
}

// ---------------------------------------------------------------------------
// Flash attention. Reference computes out[i] = sum_{j<=i} softmax(K_i.Q_j/8) V_j
// (K rows play the query role). One wave per (b,h,16-row tile), online softmax
// over 32-column steps. V tile (64x32) is staged into LDS with async
// global->LDS copies overlapped with the score WMMAs + softmax VALU work.
// P re-layout C/D -> A-fragment goes through LDS.
// ---------------------------------------------------------------------------
__global__ void attn_kernel(const u16* __restrict__ qh,
                            const u16* __restrict__ kh,
                            const u16* __restrict__ vt,
                            u16* __restrict__ ao,
                            const int* __restrict__ causal_p) {
  __shared__ __align__(16) u16 plds[8][16 * 32];   // 1KB P tile per wave
  __shared__ __align__(16) u16 vlds[8][64 * 32];   // 4KB V tile per wave
  const int lane = threadIdx.x & 31;
  const int wv   = threadIdx.x >> 5;
  const int n16  = lane & 15;
  const int hf   = lane >> 4;
  const int gw   = (blockIdx.x * blockDim.x + threadIdx.x) >> 5;
  const int itile = gw & (Ss / 16 - 1);
  const int bh    = gw >> 7;           // gw / 128
  const int h = bh & (Hh - 1);
  const int b = bh >> 4;
  const int i0 = itile * 16;
  const bool causal = (*causal_p) != 0;

  const u16* kbase = kh + ((size_t)b * Hh + h) * Ss * DH;
  const u16* qbase = qh + ((size_t)b * Hh + h) * Ss * DH;
  const u16* vbase = vt + ((size_t)b * Hh + h) * DH * Ss;

  // wave-relative LDS byte offset of this wave's V staging buffer
  // (generic LDS address keeps the LDS offset in its low 32 bits)
  const unsigned vlds_off = (unsigned)(uintptr_t)(void*)&vlds[wv][0];
  // per-lane slice for the async copy: row pair + 16B chunk
  const int arow = lane >> 2;          // 0..7 (row within group of 8)
  const int achk = (lane & 3) * 8;     // ushort offset of 16B chunk in row

  // K-row fragments (A role), held for the whole j loop. dh=64 -> 2 frags.
  BFrag ak0, ak1;
  {
    const u16* kp = kbase + (size_t)(i0 + n16) * DH;
    ak0.q[0] = *(const uint4*)(kp + hf * 8);
    ak0.q[1] = *(const uint4*)(kp + 16 + hf * 8);
    ak1.q[0] = *(const uint4*)(kp + 32 + hf * 8);
    ak1.q[1] = *(const uint4*)(kp + 48 + hf * 8);
  }

  v8f acc[4] = {};
  float rowM[8], rowL[8];
#pragma unroll
  for (int r = 0; r < 8; ++r) { rowM[r] = -1e30f; rowL[r] = 0.0f; }

  const int jend = causal ? (i0 + 16) : Ss;
  for (int j0 = 0; j0 < jend; j0 += 32) {
    // --- kick off async copy of V tile rows d=0..63, cols j0..j0+31 ------
    // (consumed only after score WMMAs + softmax; previous iteration's PV
    //  WMMA already forced DScnt==0, so no WAR hazard on the LDS buffer)
#pragma unroll
    for (int t = 0; t < 8; ++t) {
      const int row = t * 8 + arow;                       // d row 0..63
      const u16* g = vbase + (size_t)row * Ss + j0 + achk;
      async_g2l_b128(vlds_off + (unsigned)(row * 32 + achk) * 2u, g);
    }

    // --- scores: S = K_i (16 x 64) x Q_j^T (64 x 32) ---------------------
    float p0[8], p1[8];
#pragma unroll
    for (int st = 0; st < 2; ++st) {
      const int j1 = j0 + st * 16;
      BFrag bq0, bq1;
      const u16* qp = qbase + (size_t)(j1 + n16) * DH;
      bq0.q[0] = *(const uint4*)(qp + hf * 8);
      bq0.q[1] = *(const uint4*)(qp + 16 + hf * 8);
      bq1.q[0] = *(const uint4*)(qp + 32 + hf * 8);
      bq1.q[1] = *(const uint4*)(qp + 48 + hf * 8);
      v8f s = {};
      s = wmma_bf16(ak0.v, bq0.v, s);
      s = wmma_bf16(ak1.v, bq1.v, s);
      float* p = st ? p1 : p0;
#pragma unroll
      for (int r = 0; r < 8; ++r) {
        float sv = s[r] * 0.125f;                      // 1/sqrt(64)
        const int jcol = j1 + n16;
        const int irow = i0 + r + 8 * hf;
        if (causal && (jcol > irow)) sv = -1e30f;
        p[r] = sv;
      }
    }

    // --- online softmax update (per row: max, rescale, exp, sum) ---------
#pragma unroll
    for (int r = 0; r < 8; ++r) {
      float t  = hmax16(fmaxf(p0[r], p1[r]));
      float nm = fmaxf(rowM[r], t);
      float sc = __expf(rowM[r] - nm);
      rowM[r] = nm;
      p0[r] = __expf(p0[r] - nm);
      p1[r] = __expf(p1[r] - nm);
      rowL[r] = rowL[r] * sc + hsum16(p0[r] + p1[r]);
#pragma unroll
      for (int dt = 0; dt < 4; ++dt) acc[dt][r] *= sc;
      // stage P into LDS in row-major 16x32 (C/D layout -> memory)
      plds[wv][(r + 8 * hf) * 32 + n16]      = f32_to_bf16(p0[r]);
      plds[wv][(r + 8 * hf) * 32 + 16 + n16] = f32_to_bf16(p1[r]);
    }

    // reload P as an A-fragment (LDS ops are in-order within a wave)
    BFrag pf;
    const uint4* pp = (const uint4*)&plds[wv][0];
    pf.q[0] = pp[n16 * 4 + hf];
    pf.q[1] = pp[n16 * 4 + 2 + hf];

    // --- wait for async V tile, then O += P (16x32) x V (32x64) ----------
    wait_async0();
    const uint4* vp = (const uint4*)&vlds[wv][0];
#pragma unroll
    for (int dt = 0; dt < 4; ++dt) {
      BFrag vf;
      const int d = dt * 16 + n16;              // V row in LDS tile
      vf.q[0] = vp[d * 4 + hf];
      vf.q[1] = vp[d * 4 + 2 + hf];
      acc[dt] = wmma_bf16(pf.v, vf.v, acc[dt]);
    }
  }

  // normalize and store attention output [b,s,D] bf16 (token-major)
#pragma unroll
  for (int r = 0; r < 8; ++r) {
    const float inv = 1.0f / rowL[r];
    const int i = i0 + r + 8 * hf;
    const size_t base = ((size_t)b * Ss + i) * Dd + h * DH;
#pragma unroll
    for (int dt = 0; dt < 4; ++dt)
      ao[base + dt * 16 + n16] = f32_to_bf16(acc[dt][r] * inv);
  }
}

// ---------------------------------------------------------------------------
// Output projection: out = attn @ w_out^T + b_out (f32 output)
// ---------------------------------------------------------------------------
__global__ void gemm_out_kernel(const u16* __restrict__ ab,
                                const u16* __restrict__ wob,
                                const float* __restrict__ b_out,
                                float* __restrict__ out) {
  const int lane = threadIdx.x & 31;
  const int n16  = lane & 15;
  const int hf   = lane >> 4;
  const int gw   = (blockIdx.x * blockDim.x + threadIdx.x) >> 5;
  const int mstrip = gw & 127;   // 128 strips of 32 rows
  const int nstrip = gw >> 7;    // 16 strips of 64 cols
  const int m0 = mstrip * 32, n0 = nstrip * 64;

  v8f acc[2][4] = {};
#pragma unroll 4
  for (int k0 = 0; k0 < Dd; k0 += 32) {
    BFrag a0, a1;
    const u16* ap = ab + (size_t)(m0 + n16) * Dd + k0;
    a0.q[0] = *(const uint4*)(ap + hf * 8);
    a0.q[1] = *(const uint4*)(ap + 16 + hf * 8);
    const u16* ap1 = ap + (size_t)16 * Dd;
    a1.q[0] = *(const uint4*)(ap1 + hf * 8);
    a1.q[1] = *(const uint4*)(ap1 + 16 + hf * 8);
#pragma unroll
    for (int nt = 0; nt < 4; ++nt) {
      BFrag bf;
      const u16* bp = wob + (size_t)(n0 + nt * 16 + n16) * Dd + k0;
      bf.q[0] = *(const uint4*)(bp + hf * 8);
      bf.q[1] = *(const uint4*)(bp + 16 + hf * 8);
      acc[0][nt] = wmma_bf16(a0.v, bf.v, acc[0][nt]);
      acc[1][nt] = wmma_bf16(a1.v, bf.v, acc[1][nt]);
    }
  }

#pragma unroll
  for (int mi = 0; mi < 2; ++mi)
#pragma unroll
    for (int nt = 0; nt < 4; ++nt) {
      const int col = n0 + nt * 16 + n16;
      const float bias = b_out[col];
#pragma unroll
      for (int r = 0; r < 8; ++r) {
        const int row = m0 + mi * 16 + r + 8 * hf;
        out[(size_t)row * Dd + col] = acc[mi][nt][r] + bias;
      }
    }
}

// ---------------------------------------------------------------------------
// Launch
// ---------------------------------------------------------------------------
extern "C" void kernel_launch(void* const* d_in, const int* in_sizes, int n_in,
                              void* d_out, int out_size, void* d_ws, size_t ws_size,
                              hipStream_t stream) {
  (void)in_sizes; (void)n_in; (void)out_size; (void)ws_size;
  const float* x     = (const float*)d_in[0];
  const float* w_in  = (const float*)d_in[1];
  const float* b_in  = (const float*)d_in[2];
  const float* w_out = (const float*)d_in[3];
  const float* b_out = (const float*)d_in[4];
  const int*   causal = (const int*)d_in[5];
  float* out = (float*)d_out;

  // workspace layout (u16 elements)
  u16* ws  = (u16*)d_ws;
  u16* xb  = ws + 0;          // x bf16           [B*S*D]      4,194,304
  u16* wib = ws + 4194304;    // w_in bf16        [3D*D]       3,145,728
  u16* wob = ws + 7340032;    // w_out bf16       [D*D]        1,048,576
  u16* qh  = ws + 8388608;    // q head-major     [B,H,S,dh]   4,194,304
  u16* kh  = ws + 12582912;   // k head-major     [B,H,S,dh]   4,194,304
  u16* vtp = ws + 16777216;   // v transposed     [B,H,dh,S]   4,194,304
  u16* ao  = ws + 20971520;   // attn out bf16    [B,S,D]      4,194,304

  cvt_bf16_kernel<<<512, 256, 0, stream>>>((const float4*)x,    (ushort4*)xb,  4194304 / 4);
  cvt_bf16_kernel<<<512, 256, 0, stream>>>((const float4*)w_in, (ushort4*)wib, 3145728 / 4);
  cvt_bf16_kernel<<<256, 256, 0, stream>>>((const float4*)w_out,(ushort4*)wob, 1048576 / 4);

  // QKV GEMM: 128 m-strips * 48 n-strips = 6144 waves = 768 blocks of 8 waves
  gemm_qkv_kernel<<<768, 256, 0, stream>>>(xb, wib, b_in, qh, kh, vtp);

  // attention: B*H*(S/16) = 4096 waves = 512 blocks
  attn_kernel<<<512, 256, 0, stream>>>(qh, kh, vtp, ao, causal);

  // output GEMM: 128 * 16 = 2048 waves = 256 blocks
  gemm_out_kernel<<<256, 256, 0, stream>>>(ao, wob, b_out, out);
}
